// MelSpectrogramCalculator_5626407158114
// MI455X (gfx1250) — matrix operand order repaired and checked by reading, not compiled, not run on hardware
//
#include <hip/hip_runtime.h>
#include <hip/hip_bf16.h>

typedef _Float16 h8  __attribute__((ext_vector_type(8)));
typedef _Float16 h16 __attribute__((ext_vector_type(16)));
typedef float    f8  __attribute__((ext_vector_type(8)));

#define SR_LEN   160000
#define BATCH    32
#define NFRAMES  626
#define MROWS    (BATCH * NFRAMES)   // 20032 (divisible by 64)
#define WIN      1024
#define HOP      256
#define PADW     512
#define NSTFT    513
#define NPAD     544                 // 513 padded up to multiple of 32
#define NMEL     80

static __device__ __forceinline__ h16 cat8(h8 lo, h8 hi) {
  return __builtin_shufflevector(lo, hi, 0,1,2,3,4,5,6,7,8,9,10,11,12,13,14,15);
}

// ---------------------------------------------------------------------------
// 1a. frame + reflect pad + Hann window + f32->f16. Aw: (MROWS x WIN) f16.
// ---------------------------------------------------------------------------
__global__ void k_frame_window(const float* __restrict__ wav,
                               const float* __restrict__ window,
                               _Float16* __restrict__ Aw) {
  int gid = blockIdx.x * blockDim.x + threadIdx.x;   // MROWS * 128 threads
  int row = gid >> 7;
  int w0  = (gid & 127) << 3;
  if (row >= MROWS) return;
  int b  = row / NFRAMES;
  int fr = row - b * NFRAMES;
  const float* wv = wav + (size_t)b * SR_LEN;
  int base = fr * HOP - PADW + w0;
  h8 v;
#pragma unroll
  for (int i = 0; i < 8; ++i) {
    int pos = base + i;
    pos = (pos < 0) ? -pos : pos;
    pos = (pos >= SR_LEN) ? (2 * SR_LEN - 2 - pos) : pos;
    v[i] = (_Float16)(wv[pos] * window[w0 + i]);
  }
  *(h8*)(Aw + (size_t)row * WIN + w0) = v;
}

// ---------------------------------------------------------------------------
// 1b. DFT bases -> f16, zero-pad freq bins 513 -> 544 (freq-major layout).
// ---------------------------------------------------------------------------
__global__ void k_prep_basis(const float* __restrict__ rr,
                             const float* __restrict__ ri,
                             _Float16* __restrict__ Br,
                             _Float16* __restrict__ Bi) {
  int gid = blockIdx.x * blockDim.x + threadIdx.x;   // NPAD * 128 threads
  int n  = gid >> 7;
  int w0 = (gid & 127) << 3;
  if (n >= NPAD) return;
  h8 vr, vi;
  if (n < NSTFT) {
    const float* pr = rr + (size_t)n * WIN + w0;
    const float* pi = ri + (size_t)n * WIN + w0;
#pragma unroll
    for (int i = 0; i < 8; ++i) { vr[i] = (_Float16)pr[i]; vi[i] = (_Float16)pi[i]; }
  } else {
#pragma unroll
    for (int i = 0; i < 8; ++i) { vr[i] = (_Float16)0.f; vi[i] = (_Float16)0.f; }
  }
  *(h8*)(Br + (size_t)n * WIN + w0) = vr;
  *(h8*)(Bi + (size_t)n * WIN + w0) = vi;
}

// ---------------------------------------------------------------------------
// 1c. mel fbanks (513 x 80) -> transposed f16 FbT (80 x 544), zero padded.
// ---------------------------------------------------------------------------
__global__ void k_prep_fb(const float* __restrict__ fb,
                          _Float16* __restrict__ FbT) {
  int gid = blockIdx.x * blockDim.x + threadIdx.x;
  if (gid >= NMEL * NPAD) return;
  int m = gid / NPAD;
  int k = gid - m * NPAD;
  FbT[gid] = (k < NSTFT) ? (_Float16)fb[(size_t)k * NMEL + m] : (_Float16)0.f;
}

// ---------------------------------------------------------------------------
// 2. DFT via WMMA + magnitude. One wave per 32x32 macro-tile (2M x 2N
//    register blocking): 6 fragment loads -> 8 WMMAs per K-step.
// ---------------------------------------------------------------------------
__global__ void k_dft_mag(const _Float16* __restrict__ Aw,
                          const _Float16* __restrict__ Br,
                          const _Float16* __restrict__ Bi,
                          _Float16* __restrict__ Mag) {
  const int NT2 = NPAD / 32;                         // 17 macro N-tiles
  const int MT2 = MROWS / 32;                        // 626 macro M-tiles
  int wave = (blockIdx.x * blockDim.x + threadIdx.x) >> 5;
  int lane = threadIdx.x & 31;
  int mt = wave / NT2;
  int nt = wave - mt * NT2;
  if (mt >= MT2) return;                             // wave-uniform
  int row0 = mt * 32, col0 = nt * 32;
  int hlo  = lane & 15;
  int hi16 = lane >> 4;                              // 0 | 1 half-wave
  const _Float16* aB[2];
  const _Float16* brB[2];
  const _Float16* biB[2];
#pragma unroll
  for (int i = 0; i < 2; ++i) {
    aB[i]  = Aw + (size_t)(row0 + 16 * i + hlo) * WIN + (hi16 ? 16 : 0);
    brB[i] = Br + (size_t)(col0 + 16 * i + hlo) * WIN + (hi16 ? 16 : 0);
    biB[i] = Bi + (size_t)(col0 + 16 * i + hlo) * WIN + (hi16 ? 16 : 0);
  }
  f8 accR[2][2], accI[2][2];
#pragma unroll
  for (int mi = 0; mi < 2; ++mi)
#pragma unroll
    for (int ni = 0; ni < 2; ++ni)
#pragma unroll
      for (int r = 0; r < 8; ++r) { accR[mi][ni][r] = 0.f; accI[mi][ni][r] = 0.f; }

#pragma unroll 2
  for (int k0 = 0; k0 < WIN; k0 += 32) {
    h16 a[2], br[2], bi[2];
#pragma unroll
    for (int i = 0; i < 2; ++i) {
      h8 a0 = *(const h8*)(aB[i] + k0);
      h8 a1 = *(const h8*)(aB[i] + k0 + 8);
      a[i]  = cat8(a0, a1);
      br[i] = *(const h16*)(brB[i] + k0);
      bi[i] = *(const h16*)(biB[i] + k0);
    }
#pragma unroll
    for (int mi = 0; mi < 2; ++mi)
#pragma unroll
      for (int ni = 0; ni < 2; ++ni) {
        accR[mi][ni] = __builtin_amdgcn_wmma_f32_16x16x32_f16(
            false, a[mi], false, br[ni], (short)0, accR[mi][ni], false, false);
        accI[mi][ni] = __builtin_amdgcn_wmma_f32_16x16x32_f16(
            false, a[mi], false, bi[ni], (short)0, accI[mi][ni], false, false);
      }
  }
  // D layout: VGPR r -> M = r + (hi16?8:0), N = hlo (within each 16x16 tile)
#pragma unroll
  for (int mi = 0; mi < 2; ++mi) {
    int mOff = row0 + 16 * mi + (hi16 ? 8 : 0);
#pragma unroll
    for (int ni = 0; ni < 2; ++ni) {
      int n = col0 + 16 * ni + hlo;
#pragma unroll
      for (int r = 0; r < 8; ++r) {
        float re = accR[mi][ni][r], im = accI[mi][ni][r];
        Mag[(size_t)(mOff + r) * NPAD + n] = (_Float16)sqrtf(re * re + im * im);
      }
    }
  }
}

// ---------------------------------------------------------------------------
// 3. Mel projection via WMMA (4M x 1N register blocking: B-frag reused 4x),
//    fused (b, mel, frame) transpose on the f32 store.
// ---------------------------------------------------------------------------
__global__ void k_mel(const _Float16* __restrict__ Mag,
                      const _Float16* __restrict__ FbT,
                      float* __restrict__ out) {
  const int NT  = NMEL / 16;                         // 5 N-tiles
  const int MT4 = MROWS / 64;                        // 313 macro M-tiles
  int wave = (blockIdx.x * blockDim.x + threadIdx.x) >> 5;
  int lane = threadIdx.x & 31;
  int mt = wave / NT;
  int nt = wave - mt * NT;
  if (mt >= MT4) return;                             // wave-uniform
  int row0 = mt * 64, col0 = nt * 16;
  int hlo  = lane & 15;
  int hi16 = lane >> 4;
  const _Float16* aB[4];
#pragma unroll
  for (int i = 0; i < 4; ++i)
    aB[i] = Mag + (size_t)(row0 + 16 * i + hlo) * NPAD + (hi16 ? 16 : 0);
  const _Float16* bBase = FbT + (size_t)(col0 + hlo) * NPAD + (hi16 ? 16 : 0);
  f8 acc[4];
#pragma unroll
  for (int i = 0; i < 4; ++i)
#pragma unroll
    for (int r = 0; r < 8; ++r) acc[i][r] = 0.f;

#pragma unroll 2
  for (int k0 = 0; k0 < NPAD; k0 += 32) {            // 17 WMMA steps / tile
    h16 b = *(const h16*)(bBase + k0);
#pragma unroll
    for (int i = 0; i < 4; ++i) {
      h8 a0 = *(const h8*)(aB[i] + k0);
      h8 a1 = *(const h8*)(aB[i] + k0 + 8);
      h16 a = cat8(a0, a1);
      acc[i] = __builtin_amdgcn_wmma_f32_16x16x32_f16(
          false, a, false, b, (short)0, acc[i], false, false);
    }
  }
  int m = col0 + hlo;                                // mel index (all < 80)
#pragma unroll
  for (int i = 0; i < 4; ++i) {
    int mOff = row0 + 16 * i + (hi16 ? 8 : 0);
#pragma unroll
    for (int r = 0; r < 8; ++r) {
      int row = mOff + r;
      int b  = row / NFRAMES;
      int fr = row - b * NFRAMES;
      out[((size_t)b * NMEL + m) * NFRAMES + fr] = acc[i][r];
    }
  }
}

// ---------------------------------------------------------------------------
extern "C" void kernel_launch(void* const* d_in, const int* in_sizes, int n_in,
                              void* d_out, int out_size, void* d_ws, size_t ws_size,
                              hipStream_t stream) {
  const float* wav    = (const float*)d_in[0];  // (32,1,160000)
  const float* window = (const float*)d_in[1];  // (1024)
  const float* fb     = (const float*)d_in[2];  // (513,80)
  const float* rr     = (const float*)d_in[3];  // (513,1024)
  const float* ri     = (const float*)d_in[4];  // (513,1024)
  float* out = (float*)d_out;                   // (32,80,626)

  char* ws = (char*)d_ws;
  size_t szAw = (size_t)MROWS * WIN * sizeof(_Float16);   // 41,025,536
  size_t szB  = (size_t)NPAD * WIN * sizeof(_Float16);    //  1,114,112
  size_t szFb = (size_t)NMEL * NPAD * sizeof(_Float16);   //     87,040
  _Float16* Aw  = (_Float16*)(ws);
  _Float16* Brr = (_Float16*)(ws + szAw);
  _Float16* Bii = (_Float16*)(ws + szAw + szB);
  _Float16* FbT = (_Float16*)(ws + szAw + 2 * szB);
  _Float16* Mag = (_Float16*)(ws + szAw + 2 * szB + szFb); // 21,794,816 B
  // total workspace ~65.1 MB

  // 1. prep
  k_frame_window<<<(MROWS * 128) / 256, 256, 0, stream>>>(wav, window, Aw);
  k_prep_basis<<<(NPAD * 128) / 256, 256, 0, stream>>>(rr, ri, Brr, Bii);
  k_prep_fb<<<(NMEL * NPAD + 255) / 256, 256, 0, stream>>>(fb, FbT);

  // 2. DFT + magnitude: 626*17 = 10642 waves, 8 waves/block -> 1331 blocks
  k_dft_mag<<<((MROWS / 32) * (NPAD / 32) + 7) / 8, 256, 0, stream>>>(Aw, Brr, Bii, Mag);

  // 3. mel + transpose: 313*5 = 1565 waves -> 196 blocks
  k_mel<<<((MROWS / 64) * (NMEL / 16) + 7) / 8, 256, 0, stream>>>(Mag, FbT, out);
}
